// GRevHalfLayer_80625126081259
// MI455X (gfx1250) — compile-verified, matrix operand order checked by reading, not run on hardware
//
#include <hip/hip_runtime.h>
#include <hip/hip_bf16.h>
#include <math.h>

// ---------------------------------------------------------------------------
// GRevHalfLayer for MI455X (gfx1250): bf16 WMMA MLPs, L2-resident scatter-add.
// ---------------------------------------------------------------------------

typedef __attribute__((ext_vector_type(16))) __bf16 v16bf;
typedef __attribute__((ext_vector_type(8)))  float  v8f;

#define NN   50000
#define EE   800000
// transposed-bf16 weight arena layout (element offsets), per GNN:
#define GOFF 106496
#define EW0  0        // edge w0^T : 128 x 192
#define EW1  24576    // edge w1^T : 128 x 128
#define EW2  40960    // edge w2^T : 128 x 128
#define NW0  57344    // node w0^T : 128 x 192
#define NW1  81920    // node w1^T : 128 x 128
#define NW2  98304    // node w2^T :  64 x 128
// LDS row strides (elements)
#define XS   208      // 192-wide input tile + pad
#define HS   136      // 128-wide hidden tile + pad
#define SS   72       // 64-wide f32 scale tile + pad

struct Frag { union { uint4 u[2]; v16bf v; }; };

// 16-bit A/B fragment layout (ISA 7.12.2): lane l<16 -> row l, K = k0..k0+7 and
// k0+16..k0+23 ; lane l+16 -> row l-16, K = k0+8..k0+15 and k0+24..k0+31.
__device__ __forceinline__ Frag ld_frag(const __bf16* base, int stride,
                                        int rowBase, int k0, int lane) {
  const __bf16* p = base + (size_t)(rowBase + (lane & 15)) * stride
                         + k0 + ((lane >> 4) << 3);
  Frag f;
  f.u[0] = *(const uint4*)(p);
  f.u[1] = *(const uint4*)(p + 16);
  return f;
}

__device__ __forceinline__ v8f wmma_bf16(const Frag& a, const Frag& b, v8f c) {
  return __builtin_amdgcn_wmma_f32_16x16x32_bf16(
      /*neg_a=*/false, a.v, /*neg_b=*/false, b.v,
      /*c_mod=*/(short)0, c, /*reuse_a=*/false, /*reuse_b=*/false);
}

// One dense layer: Y[128 x NT*16] = act(X[128 x KT*32] @ W + b), per-wave
// 16-row slice, A fragments fully hoisted (so dst may alias src).
// ACT: 0 = none, 1 = relu, 2 = tanh
template<int KT, int NT, int ACT>
__device__ __forceinline__ void mlp_layer(const __bf16* src, int sstride,
                                          __bf16* dst, int dstride,
                                          const __bf16* __restrict__ wT,
                                          const float* __restrict__ bias,
                                          int lane, int wave) {
  Frag a[KT];
#pragma unroll
  for (int i = 0; i < KT; ++i) a[i] = ld_frag(src, sstride, wave * 16, i * 32, lane);
  const int mb = wave * 16 + ((lane >> 4) << 3);
  const int nl = lane & 15;
#pragma unroll
  for (int nt = 0; nt < NT; ++nt) {
    v8f c = {0.f, 0.f, 0.f, 0.f, 0.f, 0.f, 0.f, 0.f};
#pragma unroll
    for (int i = 0; i < KT; ++i) {
      Frag b = ld_frag(wT, KT * 32, nt * 16, i * 32, lane);
      c = wmma_bf16(a[i], b, c);
    }
    const int n = nt * 16 + nl;
    const float bv = bias[n];
#pragma unroll
    for (int i = 0; i < 8; ++i) {            // C/D layout: M = mb+i, N = n
      float v = c[i] + bv;
      if (ACT == 1) v = fmaxf(v, 0.0f);
      if (ACT == 2) v = tanhf(v);
      dst[(mb + i) * dstride + n] = (__bf16)v;
    }
  }
}

// ---------------------------------------------------------------------------
// Prep kernels
// ---------------------------------------------------------------------------
__global__ void zero_kernel(float* __restrict__ a, float* __restrict__ b,
                            float* __restrict__ ld, int n) {
  int i = blockIdx.x * blockDim.x + threadIdx.x;
  if (i == 0) *ld = 0.0f;
  int stride = gridDim.x * blockDim.x;
  for (; i < n; i += stride) { a[i] = 0.0f; b[i] = 0.0f; }
}

__global__ void cvt_half1_kernel(const float* __restrict__ nodes,
                                 __bf16* __restrict__ dst) {
  int i = blockIdx.x * blockDim.x + threadIdx.x;
  if (i < NN * 64) {
    int node = i >> 6, c = i & 63;
    dst[i] = (__bf16)nodes[(size_t)node * 128 + 64 + c];
  }
}

// src: f32 row-major K x N  ->  dst: bf16 row-major N x K
__global__ void transpose_kernel(const float* __restrict__ src,
                                 __bf16* __restrict__ dst, int K, int Nn) {
  int i = blockIdx.x * blockDim.x + threadIdx.x;
  if (i < K * Nn) {
    int k = i / Nn, n = i % Nn;
    dst[n * K + k] = (__bf16)src[i];
  }
}

// ---------------------------------------------------------------------------
// Edge kernel: 128 edges/block; both GNN edge-MLPs over one shared input tile;
// final layer scatter-adds into L2-resident agg buffers.
// ---------------------------------------------------------------------------
__global__ __launch_bounds__(256) void edge_kernel(
    const float* __restrict__ ef, const int* __restrict__ snd,
    const int* __restrict__ rcv, const __bf16* __restrict__ half1bf,
    const __bf16* __restrict__ wT,
    const float* __restrict__ b0s, const float* __restrict__ b1s,
    const float* __restrict__ b2s, const float* __restrict__ b0t,
    const float* __restrict__ b1t, const float* __restrict__ b2t,
    float* __restrict__ aggS, float* __restrict__ aggT) {
  __shared__ __align__(16) __bf16 Xs[128 * XS];
  __shared__ __align__(16) __bf16 Hs[128 * HS];
  const int t = threadIdx.x;
  const int eBase = blockIdx.x * 128;

  // stage edge_feats (f32 -> bf16): 128 rows x 16 float4
#pragma unroll
  for (int it = 0; it < 8; ++it) {
    int idx = t + it * 256;
    int r = idx >> 4, c4 = idx & 15;
    float4 f = *(const float4*)&ef[(size_t)(eBase + r) * 64 + c4 * 4];
    __bf16* dx = &Xs[r * XS + c4 * 4];
    dx[0] = (__bf16)f.x; dx[1] = (__bf16)f.y;
    dx[2] = (__bf16)f.z; dx[3] = (__bf16)f.w;
  }
  // gather half1[senders], half1[receivers] (bf16 rows, 16B chunks)
#pragma unroll
  for (int it = 0; it < 4; ++it) {
    int idx = t + it * 256;
    int r = idx >> 3, j = idx & 7;
    int s = snd[eBase + r];
    *(uint4*)&Xs[r * XS + 64 + j * 8] =
        *(const uint4*)&half1bf[(size_t)s * 64 + j * 8];
    int rr = rcv[eBase + r];
    *(uint4*)&Xs[r * XS + 128 + j * 8] =
        *(const uint4*)&half1bf[(size_t)rr * 64 + j * 8];
  }
  __syncthreads();

  const int lane = t & 31, wave = t >> 5;
  const int mb = wave * 16 + ((lane >> 4) << 3);
  const int nl = lane & 15;
#pragma unroll 1
  for (int g = 0; g < 2; ++g) {
    const __bf16* wg = wT + g * GOFF;
    const float* bb0 = g ? b0t : b0s;
    const float* bb1 = g ? b1t : b1s;
    const float* bb2 = g ? b2t : b2s;
    float* agg = g ? aggT : aggS;
    mlp_layer<6, 8, 1>(Xs, XS, Hs, HS, wg + EW0, bb0, lane, wave);   // 192->128 relu
    mlp_layer<4, 8, 1>(Hs, HS, Hs, HS, wg + EW1, bb1, lane, wave);   // 128->128 relu (in place)
    // 128->128 linear + scatter-add by receiver (agg lives in L2)
    Frag a[4];
#pragma unroll
    for (int i = 0; i < 4; ++i) a[i] = ld_frag(Hs, HS, wave * 16, i * 32, lane);
#pragma unroll
    for (int nt = 0; nt < 8; ++nt) {
      v8f c = {0.f, 0.f, 0.f, 0.f, 0.f, 0.f, 0.f, 0.f};
#pragma unroll
      for (int i = 0; i < 4; ++i) {
        Frag b = ld_frag(wg + EW2, 128, nt * 16, i * 32, lane);
        c = wmma_bf16(a[i], b, c);
      }
      const int n = nt * 16 + nl;
      const float bv = bb2[n];
#pragma unroll
      for (int i = 0; i < 8; ++i) {
        int e = eBase + mb + i;
        int r = rcv[e];
        atomicAdd(&agg[(size_t)r * 128 + n], c[i] + bv);
      }
    }
  }
}

// ---------------------------------------------------------------------------
// Node kernel: 128 nodes/block; scale MLP (tanh) -> LDS, trans MLP (relu),
// fused half0*exp(scale)+trans, half1 passthrough, logdetJ reduction.
// ---------------------------------------------------------------------------
__global__ __launch_bounds__(256) void node_kernel(
    const float* __restrict__ nodes, const __bf16* __restrict__ wT,
    const float* __restrict__ b0s, const float* __restrict__ b1s,
    const float* __restrict__ b2s, const float* __restrict__ b0t,
    const float* __restrict__ b1t, const float* __restrict__ b2t,
    const float* __restrict__ aggS, const float* __restrict__ aggT,
    float* __restrict__ out, float* __restrict__ logdet) {
  __shared__ __align__(16) __bf16 Xs[128 * XS];
  __shared__ __align__(16) __bf16 Hs[128 * HS];
  __shared__ __align__(16) float  Ss[128 * SS];
  const int t = threadIdx.x;
  const int nBase = blockIdx.x * 128;

  // stage half1 -> X cols [0,64) and passthrough copy to out cols [64,128)
#pragma unroll
  for (int it = 0; it < 8; ++it) {
    int idx = t + it * 256;
    int r = idx >> 4, c4 = idx & 15;
    int node = nBase + r;
    int cn = node < NN ? node : NN - 1;         // clamp (WMMA needs full EXEC)
    float4 f = *(const float4*)&nodes[(size_t)cn * 128 + 64 + c4 * 4];
    __bf16* dx = &Xs[r * XS + c4 * 4];
    dx[0] = (__bf16)f.x; dx[1] = (__bf16)f.y;
    dx[2] = (__bf16)f.z; dx[3] = (__bf16)f.w;
    if (node < NN) *(float4*)&out[(size_t)node * 128 + 64 + c4 * 4] = f;
  }
  const int lane = t & 31, wave = t >> 5;
  const int mb = wave * 16 + ((lane >> 4) << 3);
  const int nl = lane & 15;
  float lsum = 0.0f;

  // ---- GNN 0: scale (tanh hidden activations) ----
  __syncthreads();
#pragma unroll
  for (int it = 0; it < 16; ++it) {             // agg -> X cols [64,192)
    int idx = t + it * 256;
    int r = idx >> 5, c4 = idx & 31;
    int node = nBase + r;
    int cn = node < NN ? node : NN - 1;
    float4 f = *(const float4*)&aggS[(size_t)cn * 128 + c4 * 4];
    __bf16* dx = &Xs[r * XS + 64 + c4 * 4];
    dx[0] = (__bf16)f.x; dx[1] = (__bf16)f.y;
    dx[2] = (__bf16)f.z; dx[3] = (__bf16)f.w;
  }
  __syncthreads();
  {
    const __bf16* wg = wT;                      // scale GNN arena
    mlp_layer<6, 8, 2>(Xs, XS, Hs, HS, wg + NW0, b0s, lane, wave);
    mlp_layer<4, 8, 2>(Hs, HS, Hs, HS, wg + NW1, b1s, lane, wave);
    Frag a[4];
#pragma unroll
    for (int i = 0; i < 4; ++i) a[i] = ld_frag(Hs, HS, wave * 16, i * 32, lane);
#pragma unroll
    for (int nt = 0; nt < 4; ++nt) {            // 128 -> 64, linear
      v8f c = {0.f, 0.f, 0.f, 0.f, 0.f, 0.f, 0.f, 0.f};
#pragma unroll
      for (int i = 0; i < 4; ++i) {
        Frag b = ld_frag(wg + NW2, 128, nt * 16, i * 32, lane);
        c = wmma_bf16(a[i], b, c);
      }
      const int n = nt * 16 + nl;
      const float bv = b2s[n];
#pragma unroll
      for (int i = 0; i < 8; ++i) {
        float v = c[i] + bv;
        Ss[(mb + i) * SS + n] = v;
        if (nBase + mb + i < NN) lsum += v;     // logdetJ contribution
      }
    }
  }

  // ---- GNN 1: trans (relu hidden activations) ----
  __syncthreads();
#pragma unroll
  for (int it = 0; it < 16; ++it) {
    int idx = t + it * 256;
    int r = idx >> 5, c4 = idx & 31;
    int node = nBase + r;
    int cn = node < NN ? node : NN - 1;
    float4 f = *(const float4*)&aggT[(size_t)cn * 128 + c4 * 4];
    __bf16* dx = &Xs[r * XS + 64 + c4 * 4];
    dx[0] = (__bf16)f.x; dx[1] = (__bf16)f.y;
    dx[2] = (__bf16)f.z; dx[3] = (__bf16)f.w;
  }
  __syncthreads();
  {
    const __bf16* wg = wT + GOFF;               // trans GNN arena
    mlp_layer<6, 8, 1>(Xs, XS, Hs, HS, wg + NW0, b0t, lane, wave);
    mlp_layer<4, 8, 1>(Hs, HS, Hs, HS, wg + NW1, b1t, lane, wave);
    Frag a[4];
#pragma unroll
    for (int i = 0; i < 4; ++i) a[i] = ld_frag(Hs, HS, wave * 16, i * 32, lane);
#pragma unroll
    for (int nt = 0; nt < 4; ++nt) {
      v8f c = {0.f, 0.f, 0.f, 0.f, 0.f, 0.f, 0.f, 0.f};
#pragma unroll
      for (int i = 0; i < 4; ++i) {
        Frag b = ld_frag(wg + NW2, 128, nt * 16, i * 32, lane);
        c = wmma_bf16(a[i], b, c);
      }
      const int n = nt * 16 + nl;
      const float bv = b2t[n];
#pragma unroll
      for (int i = 0; i < 8; ++i) {
        int node = nBase + mb + i;
        if (node < NN) {
          float v  = c[i] + bv;
          float sc = Ss[(mb + i) * SS + n];
          out[(size_t)node * 128 + n] =
              nodes[(size_t)node * 128 + n] * expf(sc) + v;
        }
      }
    }
  }

  // logdetJ: wave32 shuffle reduce, one atomic per wave
  float s = lsum;
#pragma unroll
  for (int off = 16; off > 0; off >>= 1) s += __shfl_down(s, off, 32);
  if (lane == 0) atomicAdd(logdet, s);
}

// ---------------------------------------------------------------------------
// Host launcher
// ---------------------------------------------------------------------------
extern "C" void kernel_launch(void* const* d_in, const int* in_sizes, int n_in,
                              void* d_out, int out_size, void* d_ws,
                              size_t ws_size, hipStream_t stream) {
  (void)in_sizes; (void)n_in; (void)out_size; (void)ws_size;
  // d_in leaf order: 0 nodes, 1 edge_feats, 2-4 scale_ew, 5-7 scale_eb,
  // 8-10 scale_nw, 11-13 scale_nb, 14-16 trans_ew, 17-19 trans_eb,
  // 20-22 trans_nw, 23-25 trans_nb, 26 senders, 27 receivers
  const float* nodes = (const float*)d_in[0];
  const float* ef    = (const float*)d_in[1];
  const int*   snd   = (const int*)d_in[26];
  const int*   rcv   = (const int*)d_in[27];

  char* ws = (char*)d_ws;
  __bf16* half1bf = (__bf16*)ws;                           // 6,400,000 B
  __bf16* wT      = (__bf16*)(ws + 6400000);               //   425,984 B
  float*  aggS    = (float*)(ws + 6400000 + 425984);       // 25.6 MB
  float*  aggT    = aggS + (size_t)NN * 128;               // 25.6 MB
  float*  out     = (float*)d_out;
  float*  logdet  = out + (size_t)NN * 128;

  zero_kernel<<<2048, 256, 0, stream>>>(aggS, aggT, logdet, NN * 128);
  cvt_half1_kernel<<<(NN * 64 + 255) / 256, 256, 0, stream>>>(nodes, half1bf);

  const int widx[12] = {2, 3, 4, 8, 9, 10, 14, 15, 16, 20, 21, 22};
  const int woff[12] = {EW0, EW1, EW2, NW0, NW1, NW2,
                        GOFF + EW0, GOFF + EW1, GOFF + EW2,
                        GOFF + NW0, GOFF + NW1, GOFF + NW2};
  const int wK[12] = {192, 128, 128, 192, 128, 128, 192, 128, 128, 192, 128, 128};
  const int wN[12] = {128, 128, 128, 128, 128,  64, 128, 128, 128, 128, 128,  64};
  for (int j = 0; j < 12; ++j) {
    int total = wK[j] * wN[j];
    transpose_kernel<<<(total + 255) / 256, 256, 0, stream>>>(
        (const float*)d_in[widx[j]], wT + woff[j], wK[j], wN[j]);
  }

  edge_kernel<<<EE / 128, 256, 0, stream>>>(
      ef, snd, rcv, half1bf, wT,
      (const float*)d_in[5],  (const float*)d_in[6],  (const float*)d_in[7],
      (const float*)d_in[17], (const float*)d_in[18], (const float*)d_in[19],
      aggS, aggT);

  node_kernel<<<(NN + 127) / 128, 256, 0, stream>>>(
      nodes, wT,
      (const float*)d_in[11], (const float*)d_in[12], (const float*)d_in[13],
      (const float*)d_in[23], (const float*)d_in[24], (const float*)d_in[25],
      aggS, aggT, out, logdet);
}
